// FlashDeltaHybrid_2156073582759
// MI455X (gfx1250) — matrix operand order repaired and checked by reading, not compiled, not run on hardware
//
#include <hip/hip_runtime.h>
#include <hip/hip_bf16.h>
#include <cstdint>

#define B_ 2
#define L_ 2048
#define D_ 1024
#define H_ 16
#define HD_ 64
#define NROWS (B_*L_)     /* 4096 */
#define D3_ (3*D_)        /* 3072 */

typedef _Float16 half_t;
typedef __attribute__((ext_vector_type(16))) _Float16 v16h;
typedef __attribute__((ext_vector_type(8)))  _Float16 v8h;
typedef __attribute__((ext_vector_type(8)))  float    v8f;

static __device__ __forceinline__ v8f wmma_f16(v16h a, v16h b, v8f c) {
  // emits v_wmma_f32_16x16x32_f16
  return __builtin_amdgcn_wmma_f32_16x16x32_f16(false, a, false, b, (short)0, c, false, false);
}

// LDS byte offset of a generic pointer to __shared__ (ISA: ADDR[31:0] is the LDS offset)
static __device__ __forceinline__ uint32_t lds_off(const void* p) {
  return (uint32_t)(uintptr_t)p;
}
static __device__ __forceinline__ void async_load_b32(uint32_t ldsoff, const void* gaddr) {
  asm volatile("global_load_async_to_lds_b32 %0, %1, off"
               :: "v"(ldsoff), "v"((uint64_t)(uintptr_t)gaddr) : "memory");
}
static __device__ __forceinline__ void wait_async0() {
  asm volatile("s_wait_asynccnt 0" ::: "memory");
}

// ---------------------------------------------------------------- elementwise
__global__ void cvt_f32_f16(const float* __restrict__ in, half_t* __restrict__ out, long n) {
  long i = (long)blockIdx.x * blockDim.x + threadIdx.x;
  long stride = (long)gridDim.x * blockDim.x;
  for (; i < n; i += stride) out[i] = (half_t)in[i];
}

__global__ void sigmoid_bias(float* __restrict__ x, const float* __restrict__ bias,
                             int lastdim, long n) {
  long i = (long)blockIdx.x * blockDim.x + threadIdx.x;
  long stride = (long)gridDim.x * blockDim.x;
  for (; i < n; i += stride) {
    float v = x[i] + bias[i % lastdim];
    x[i] = 1.0f / (1.0f + __expf(-v));
  }
}

__global__ void combine_gate(const float* __restrict__ gate, const float* __restrict__ fl,
                             const float* __restrict__ dl, float* __restrict__ out, long n) {
  long i = (long)blockIdx.x * blockDim.x + threadIdx.x;
  long stride = (long)gridDim.x * blockDim.x;
  for (; i < n; i += stride) {
    float g = gate[i];
    out[i] = g * fl[i] + (1.0f - g) * dl[i];
  }
}

// -------------------------------------------------- weight transpose (f32 KxN -> f16 NxK)
__global__ __launch_bounds__(256)
void transpose_f32_f16(const float* __restrict__ in, half_t* __restrict__ out, int K, int N) {
  __shared__ float t[32][33];
  int n0 = blockIdx.x * 32, k0 = blockIdx.y * 32;
  int tx = threadIdx.x, ty = threadIdx.y;   // 32 x 8
  #pragma unroll
  for (int i = ty; i < 32; i += 8) {
    int k = k0 + i, n = n0 + tx;
    t[i][tx] = (k < K && n < N) ? in[(size_t)k * N + n] : 0.0f;
  }
  __syncthreads();
  #pragma unroll
  for (int i = ty; i < 32; i += 8) {
    int n = n0 + i, k = k0 + tx;
    if (n < N && k < K) out[(size_t)n * K + k] = (half_t)t[tx][i];
  }
}

// -------------------------------------------- V transpose: qkvh -> vT[b][h][d][l] (f16)
__global__ __launch_bounds__(256)
void vtrans(const half_t* __restrict__ qkvh, half_t* __restrict__ vT) {
  __shared__ half_t t[32][33];
  int l0 = blockIdx.x * 32, d0 = blockIdx.y * 32;
  int bh = blockIdx.z;
  int b = bh / H_, h = bh % H_;
  int tx = threadIdx.x, ty = threadIdx.y;
  #pragma unroll
  for (int i = ty; i < 32; i += 8)
    t[i][tx] = qkvh[(size_t)(b * L_ + l0 + i) * D3_ + 2 * D_ + h * HD_ + d0 + tx];
  __syncthreads();
  #pragma unroll
  for (int i = ty; i < 32; i += 8)
    vT[((size_t)bh * HD_ + d0 + i) * L_ + l0 + tx] = t[tx][i];
}

// ------------------------------------------------------------------ WMMA GEMM
// C[M,N] f32 = A[M,K] f16 row-major  x  Bt[N,K] f16 row-major (pre-transposed weights).
// Wave computes a 16x64 strip (4 N-tiles, A fragment reused 4x). 4 waves/block.
__global__ __launch_bounds__(128)
void gemm_wmma_bt(const half_t* __restrict__ A, const half_t* __restrict__ Bt,
                  float* __restrict__ C, int M, int N, int K) {
  int lane = threadIdx.x & 31;
  int wave = threadIdx.x >> 5;
  int nbase = (blockIdx.x * 4 + wave) * 64;
  int m0 = blockIdx.y * 16;
  if (nbase >= N) return;
  int hi = lane >> 4;
  int lo = lane & 15;
  const half_t* arow = A + (size_t)(m0 + lo) * K;
  const half_t* br[4];
  #pragma unroll
  for (int t = 0; t < 4; ++t) {
    int n = nbase + 16 * t + lo;
    if (n >= N) n = N - 1;                  // clamp (stores are guarded)
    br[t] = Bt + (size_t)n * K;
  }
  v8f acc0 = {}, acc1 = {}, acc2 = {}, acc3 = {};
  for (int kk = 0; kk < K; kk += 32) {
    __builtin_prefetch(arow + kk + 256, 0, 1);
    __builtin_prefetch(br[0] + kk + 256, 0, 1);
    v16h a;
    *(v8h*)&a       = *(const v8h*)(arow + kk + hi * 16);
    *((v8h*)&a + 1) = *(const v8h*)(arow + kk + hi * 16 + 8);
    v16h b0 = *(const v16h*)(br[0] + kk + hi * 16);
    v16h b1 = *(const v16h*)(br[1] + kk + hi * 16);
    v16h b2 = *(const v16h*)(br[2] + kk + hi * 16);
    v16h b3 = *(const v16h*)(br[3] + kk + hi * 16);
    acc0 = wmma_f16(a, b0, acc0);
    acc1 = wmma_f16(a, b1, acc1);
    acc2 = wmma_f16(a, b2, acc2);
    acc3 = wmma_f16(a, b3, acc3);
  }
  float* crow = C + (size_t)(m0 + hi * 8) * N + lo;
  #pragma unroll
  for (int t = 0; t < 4; ++t) {
    int n0 = nbase + 16 * t;
    if (n0 >= N) break;
    v8f acc = (t == 0) ? acc0 : (t == 1) ? acc1 : (t == 2) ? acc2 : acc3;
    #pragma unroll
    for (int r = 0; r < 8; ++r) crow[(size_t)r * N + n0] = acc[r];
  }
}

// ------------------------------------------------------------ flash attention
// One wave per (b, h, 16-row q tile); streaming causal softmax over 32-wide k chunks.
// K fragments: contiguous from qkvh. V fragments: contiguous from vT (per-head transposed).
__global__ __launch_bounds__(32)
void flash_attn(const half_t* __restrict__ qkvh, const half_t* __restrict__ vT,
                float* __restrict__ outp) {
  int q0 = blockIdx.x * 16;
  int h  = blockIdx.y;
  int b  = blockIdx.z;
  int bh = b * H_ + h;
  int lane = threadIdx.x & 31;
  int hi = lane >> 4, lo = lane & 15;
  int rbase = hi * 8;
  __shared__ __align__(32) half_t Pld[16][32];

  v16h aq0, aq1;
  {
    const half_t* qrow = qkvh + (size_t)(b * L_ + q0 + lo) * D3_ + h * HD_;
    *(v8h*)&aq0       = *(const v8h*)(qrow + 0  + hi * 16);
    *((v8h*)&aq0 + 1) = *(const v8h*)(qrow + 8  + hi * 16);
    *(v8h*)&aq1       = *(const v8h*)(qrow + 32 + hi * 16);
    *((v8h*)&aq1 + 1) = *(const v8h*)(qrow + 40 + hi * 16);
  }
  float m[8], l[8];
  #pragma unroll
  for (int r = 0; r < 8; ++r) { m[r] = -1e30f; l[r] = 0.0f; }
  v8f o0 = {}, o1 = {}, o2 = {}, o3 = {};
  const float scale = 0.125f;  // HD^-0.5
  const half_t* vt0 = vT + ((size_t)bh * HD_ + 0  + lo) * L_;
  const half_t* vt1 = vT + ((size_t)bh * HD_ + 16 + lo) * L_;
  const half_t* vt2 = vT + ((size_t)bh * HD_ + 32 + lo) * L_;
  const half_t* vt3 = vT + ((size_t)bh * HD_ + 48 + lo) * L_;

  for (int k0 = 0; k0 < q0 + 16; k0 += 32) {
    // ---- scores
    v8f s0 = {}, s1 = {};
    {
      const half_t* kr0 = qkvh + (size_t)(b * L_ + k0 + lo) * D3_ + D_ + h * HD_ + hi * 16;
      const half_t* kr1 = kr0 + (size_t)16 * D3_;
      v16h b00 = *(const v16h*)(kr0);
      v16h b01 = *(const v16h*)(kr0 + 32);
      v16h b10 = *(const v16h*)(kr1);
      v16h b11 = *(const v16h*)(kr1 + 32);
      s0 = wmma_f16(aq0, b00, s0); s0 = wmma_f16(aq1, b01, s0);
      s1 = wmma_f16(aq0, b10, s1); s1 = wmma_f16(aq1, b11, s1);
    }
    // ---- online softmax
    #pragma unroll
    for (int r = 0; r < 8; ++r) {
      int qrow = q0 + rbase + r;
      float v0 = s0[r] * scale, v1 = s1[r] * scale;
      if (k0 + lo > qrow)      v0 = -1e30f;
      if (k0 + 16 + lo > qrow) v1 = -1e30f;
      float mx = fmaxf(v0, v1);
      mx = fmaxf(mx, __shfl_xor(mx, 1, 32));
      mx = fmaxf(mx, __shfl_xor(mx, 2, 32));
      mx = fmaxf(mx, __shfl_xor(mx, 4, 32));
      mx = fmaxf(mx, __shfl_xor(mx, 8, 32));
      float mn = fmaxf(m[r], mx);
      float sc = __expf(m[r] - mn);
      float p0 = __expf(v0 - mn);
      float p1 = __expf(v1 - mn);
      float rs = p0 + p1;
      rs += __shfl_xor(rs, 1, 32);
      rs += __shfl_xor(rs, 2, 32);
      rs += __shfl_xor(rs, 4, 32);
      rs += __shfl_xor(rs, 8, 32);
      l[r] = l[r] * sc + rs;
      m[r] = mn;
      o0[r] *= sc; o1[r] *= sc; o2[r] *= sc; o3[r] *= sc;
      Pld[rbase + r][lo]      = (half_t)p0;
      Pld[rbase + r][16 + lo] = (half_t)p1;
    }
    __syncthreads();
    v16h pf;
    *(v8h*)&pf       = *(const v8h*)(&Pld[lo][hi * 16]);
    *((v8h*)&pf + 1) = *(const v8h*)(&Pld[lo][hi * 16 + 8]);
    // ---- V fragments: contiguous 16-half loads from transposed V
    v16h vb0 = *(const v16h*)(vt0 + k0 + hi * 16);
    v16h vb1 = *(const v16h*)(vt1 + k0 + hi * 16);
    v16h vb2 = *(const v16h*)(vt2 + k0 + hi * 16);
    v16h vb3 = *(const v16h*)(vt3 + k0 + hi * 16);
    o0 = wmma_f16(pf, vb0, o0);
    o1 = wmma_f16(pf, vb1, o1);
    o2 = wmma_f16(pf, vb2, o2);
    o3 = wmma_f16(pf, vb3, o3);
    __syncthreads();
  }
  float* orow = outp + (size_t)(b * L_ + q0 + rbase) * D_ + h * HD_ + lo;
  #pragma unroll
  for (int r = 0; r < 8; ++r) {
    float inv = 1.0f / l[r];
    orow[(size_t)r * D_ + 0]  = o0[r] * inv;
    orow[(size_t)r * D_ + 16] = o1[r] * inv;
    orow[(size_t)r * D_ + 32] = o2[r] * inv;
    orow[(size_t)r * D_ + 48] = o3[r] * inv;
  }
}

// --------------------------------------------------------------- depthwise conv
__global__ void conv4(const float* __restrict__ in, const float* __restrict__ w,
                      const float* __restrict__ bias, float* __restrict__ out) {
  long i = (long)blockIdx.x * blockDim.x + threadIdx.x;   // over B*L*D
  if (i >= (long)NROWS * D_) return;
  int c = (int)(i % D_);
  long bl = i / D_;
  int l = (int)(bl % L_);
  float acc = bias[c];
  #pragma unroll
  for (int j = 0; j < 4; ++j) {
    int t = l - 3 + j;
    if (t >= 0) acc += w[c * 4 + j] * in[i + (long)(t - l) * D_];
  }
  out[i] = acc;
}

// ------------------------------------------------------------------- k normalize
__global__ void knorm(float* __restrict__ dk) {
  int i = blockIdx.x * blockDim.x + threadIdx.x;   // over NROWS*H
  if (i >= NROWS * H_) return;
  int h = i % H_;
  int row = i / H_;
  float* p = dk + (size_t)row * D_ + h * HD_;
  float s = 0.0f;
  for (int j = 0; j < HD_; ++j) { float v = p[j]; s += v * v; }
  float inv = 1.0f / fmaxf(sqrtf(s), 1e-12f);
  for (int j = 0; j < HD_; ++j) p[j] *= inv;
}

// -------------------------------------------------------------------- delta scan
// One 64-thread block per (b,h); thread i owns S row i (64 regs) and z[i].
// Per-step operands staged with async-to-LDS (ASYNCcnt), double-buffered so the
// next timestep's loads overlap the current step's FMAs.
__global__ __launch_bounds__(64)
void delta_scan(const float* __restrict__ dq, const float* __restrict__ dk,
                const float* __restrict__ dv, const float* __restrict__ beta,
                float* __restrict__ ys, float* __restrict__ outS, float* __restrict__ outz) {
  int h = blockIdx.x, b = blockIdx.y;
  int i = threadIdx.x;
  __shared__ float qs[2][64], ks[2][64], vs[2][64];
  __shared__ float bsh[2];
  __shared__ float red[64];
  float S[64];
  #pragma unroll
  for (int j = 0; j < 64; ++j) S[j] = 0.0f;
  float zi = 0.0f;

  auto issue = [&](int l, int buf) {
    size_t base = (size_t)(b * L_ + l) * D_ + h * HD_ + i;
    async_load_b32(lds_off(&qs[buf][i]), dq + base);
    async_load_b32(lds_off(&ks[buf][i]), dk + base);
    async_load_b32(lds_off(&vs[buf][i]), dv + base);
    if (i == 0)
      async_load_b32(lds_off(&bsh[buf]), beta + (size_t)(b * L_ + l) * H_ + h);
  };

  issue(0, 0);
  for (int l = 0; l < L_; ++l) {
    int cur = l & 1;
    wait_async0();
    __syncthreads();                       // buffers for step l ready everywhere
    if (l + 1 < L_) issue(l + 1, cur ^ 1); // overlap next step's loads
    float y = 0.0f, old = 0.0f;
    #pragma unroll
    for (int j = 0; j < 64; ++j) { y += S[j] * qs[cur][j]; old += S[j] * ks[cur][j]; }
    red[i] = zi * qs[cur][i];
    __syncthreads();
    float denom = 1e-6f;
    #pragma unroll
    for (int j = 0; j < 64; ++j) denom += red[j];
    ys[(size_t)(b * L_ + l) * D_ + h * HD_ + i] = y / denom;
    float bt = bsh[cur];
    float bd = bt * (vs[cur][i] - old);
    #pragma unroll
    for (int j = 0; j < 64; ++j) S[j] += bd * ks[cur][j];
    zi += bt * ks[cur][i];
  }
  float* Sp = outS + ((size_t)(b * H_ + h) * HD_ + i) * HD_;
  for (int j = 0; j < 64; ++j) Sp[j] = S[j];
  outz[(size_t)(b * H_ + h) * HD_ + i] = zi;
}

// ---------------------------------------------------------------------- layernorm
__global__ __launch_bounds__(32)
void layernorm_k(const float* __restrict__ in, float* __restrict__ out,
                 const float* __restrict__ g, const float* __restrict__ bb) {
  int row = blockIdx.x;
  int lane = threadIdx.x;
  const float* p = in + (size_t)row * D_;
  float s = 0.0f, ss = 0.0f;
  #pragma unroll
  for (int j = 0; j < D_ / 32; ++j) { float v = p[lane + j * 32]; s += v; ss += v * v; }
  #pragma unroll
  for (int o = 1; o < 32; o <<= 1) { s += __shfl_xor(s, o, 32); ss += __shfl_xor(ss, o, 32); }
  float mean = s * (1.0f / D_);
  float var  = ss * (1.0f / D_) - mean * mean;
  float rstd = rsqrtf(var + 1e-5f);
  float* q = out + (size_t)row * D_;
  #pragma unroll
  for (int j = 0; j < D_ / 32; ++j) {
    int c = lane + j * 32;
    q[c] = (p[c] - mean) * rstd * g[c] + bb[c];
  }
}

// ======================================================================= launch
extern "C" void kernel_launch(void* const* d_in, const int* in_sizes, int n_in,
                              void* d_out, int out_size, void* d_ws, size_t ws_size,
                              hipStream_t stream) {
  const float* x     = (const float*)d_in[0];
  const float* Wqkv  = (const float*)d_in[1];
  const float* Wout  = (const float*)d_in[2];
  const float* Wgate = (const float*)d_in[3];
  const float* bgate = (const float*)d_in[4];
  const float* fn_g  = (const float*)d_in[5];
  const float* fn_b  = (const float*)d_in[6];
  const float* dn_g  = (const float*)d_in[7];
  const float* dn_b  = (const float*)d_in[8];
  const float* dWq   = (const float*)d_in[9];
  const float* dWk   = (const float*)d_in[10];
  const float* dWv   = (const float*)d_in[11];
  const float* dWo   = (const float*)d_in[12];
  const float* Wbeta = (const float*)d_in[13];
  const float* bbeta = (const float*)d_in[14];
  const float* cq_w  = (const float*)d_in[15];
  const float* cq_b  = (const float*)d_in[16];
  const float* ck_w  = (const float*)d_in[17];
  const float* ck_b  = (const float*)d_in[18];
  const float* ln_g  = (const float*)d_in[19];
  const float* ln_b  = (const float*)d_in[20];

  char* ws = (char*)d_ws;
  size_t off = 0;
  auto alloc = [&](size_t bytes) -> void* {
    void* p = ws + off;
    off += (bytes + 255) & ~(size_t)255;
    return p;
  };
  // f16 buffers (weights pre-transposed to NxK)
  half_t* xh     = (half_t*)alloc((size_t)NROWS * D_ * 2);
  half_t* wqkvT  = (half_t*)alloc((size_t)D3_ * D_ * 2);
  half_t* woT    = (half_t*)alloc((size_t)D_ * D_ * 2);
  half_t* wgT    = (half_t*)alloc((size_t)D_ * D_ * 2);
  half_t* dwqT   = (half_t*)alloc((size_t)D_ * D_ * 2);
  half_t* dwkT   = (half_t*)alloc((size_t)D_ * D_ * 2);
  half_t* dwvT   = (half_t*)alloc((size_t)D_ * D_ * 2);
  half_t* dwoT   = (half_t*)alloc((size_t)D_ * D_ * 2);
  half_t* wbT    = (half_t*)alloc((size_t)H_ * D_ * 2);
  half_t* qkvh   = (half_t*)alloc((size_t)NROWS * D3_ * 2);
  half_t* vTh    = (half_t*)alloc((size_t)B_ * H_ * HD_ * L_ * 2);
  half_t* fph    = (half_t*)alloc((size_t)NROWS * D_ * 2);
  half_t* ysh    = (half_t*)alloc((size_t)NROWS * D_ * 2);
  // f32 buffers (with lifetime reuse)
  float* qkvf      = (float*)alloc((size_t)NROWS * D3_ * 4);  // later: dqpre|dkpre|dvb
  float* flashpre  = (float*)alloc((size_t)NROWS * D_ * 4);   // later: gate
  float* flashproj = (float*)alloc((size_t)NROWS * D_ * 4);
  float* dqc       = (float*)alloc((size_t)NROWS * D_ * 4);
  float* dkc       = (float*)alloc((size_t)NROWS * D_ * 4);
  float* betab     = (float*)alloc((size_t)NROWS * H_ * 4);
  float* ysf       = (float*)alloc((size_t)NROWS * D_ * 4);   // later: dproj

  float* dqpre = qkvf;
  float* dkpre = qkvf + (size_t)NROWS * D_;
  float* dvb   = qkvf + (size_t)2 * NROWS * D_;
  float* gateb = flashpre;
  float* dproj = ysf;

  float* outp = (float*)d_out;
  float* outS = outp + (size_t)NROWS * D_;
  float* outz = outS + (size_t)B_ * H_ * HD_ * HD_;

  dim3 cvtg(2048), cvtb(256);
  dim3 tb(32, 8);
  auto tgrid = [](int K, int N) { return dim3((N + 31) / 32, (K + 31) / 32); };

  // 1) x -> f16 ; weights -> transposed f16
  cvt_f32_f16<<<cvtg, cvtb, 0, stream>>>(x, xh, (long)NROWS * D_);
  transpose_f32_f16<<<tgrid(D_, D3_), tb, 0, stream>>>(Wqkv,  wqkvT, D_, D3_);
  transpose_f32_f16<<<tgrid(D_, D_),  tb, 0, stream>>>(Wout,  woT,   D_, D_);
  transpose_f32_f16<<<tgrid(D_, D_),  tb, 0, stream>>>(Wgate, wgT,   D_, D_);
  transpose_f32_f16<<<tgrid(D_, D_),  tb, 0, stream>>>(dWq,   dwqT,  D_, D_);
  transpose_f32_f16<<<tgrid(D_, D_),  tb, 0, stream>>>(dWk,   dwkT,  D_, D_);
  transpose_f32_f16<<<tgrid(D_, D_),  tb, 0, stream>>>(dWv,   dwvT,  D_, D_);
  transpose_f32_f16<<<tgrid(D_, D_),  tb, 0, stream>>>(dWo,   dwoT,  D_, D_);
  transpose_f32_f16<<<tgrid(D_, H_),  tb, 0, stream>>>(Wbeta, wbT,   D_, H_);

  // 2) qkv = x @ Wqkv ; f16 copy ; per-head transposed V
  gemm_wmma_bt<<<dim3((D3_ + 255) / 256, NROWS / 16), 128, 0, stream>>>(xh, wqkvT, qkvf, NROWS, D3_, D_);
  cvt_f32_f16<<<cvtg, cvtb, 0, stream>>>(qkvf, qkvh, (long)NROWS * D3_);
  vtrans<<<dim3(L_ / 32, HD_ / 32, B_ * H_), tb, 0, stream>>>(qkvh, vTh);

  // 3) delta pre-projections (reuse qkvf region) + beta
  gemm_wmma_bt<<<dim3((D_ + 255) / 256, NROWS / 16), 128, 0, stream>>>(xh, dwqT, dqpre, NROWS, D_, D_);
  gemm_wmma_bt<<<dim3((D_ + 255) / 256, NROWS / 16), 128, 0, stream>>>(xh, dwkT, dkpre, NROWS, D_, D_);
  gemm_wmma_bt<<<dim3((D_ + 255) / 256, NROWS / 16), 128, 0, stream>>>(xh, dwvT, dvb,   NROWS, D_, D_);
  gemm_wmma_bt<<<dim3(1, NROWS / 16), 128, 0, stream>>>(xh, wbT, betab, NROWS, H_, D_);
  sigmoid_bias<<<cvtg, cvtb, 0, stream>>>(betab, bbeta, H_, (long)NROWS * H_);

  // 4) flash attention ; project ; layernorm (fn)
  flash_attn<<<dim3(L_ / 16, H_, B_), 32, 0, stream>>>(qkvh, vTh, flashpre);
  cvt_f32_f16<<<cvtg, cvtb, 0, stream>>>(flashpre, fph, (long)NROWS * D_);
  gemm_wmma_bt<<<dim3((D_ + 255) / 256, NROWS / 16), 128, 0, stream>>>(fph, woT, flashproj, NROWS, D_, D_);
  layernorm_k<<<NROWS, 32, 0, stream>>>(flashproj, flashproj, fn_g, fn_b);

  // 5) gate = sigmoid(x@Wgate + bgate)  (reuses flashpre region, now dead)
  gemm_wmma_bt<<<dim3((D_ + 255) / 256, NROWS / 16), 128, 0, stream>>>(xh, wgT, gateb, NROWS, D_, D_);
  sigmoid_bias<<<cvtg, cvtb, 0, stream>>>(gateb, bgate, D_, (long)NROWS * D_);

  // 6) causal depthwise convs + k-normalization
  conv4<<<(NROWS * D_) / 256, 256, 0, stream>>>(dqpre, cq_w, cq_b, dqc);
  conv4<<<(NROWS * D_) / 256, 256, 0, stream>>>(dkpre, ck_w, ck_b, dkc);
  knorm<<<(NROWS * H_) / 256, 256, 0, stream>>>(dkc);

  // 7) sequential delta scan -> ys, and S/z directly into d_out
  delta_scan<<<dim3(H_, B_), 64, 0, stream>>>(dqc, dkc, dvb, betab, ysf, outS, outz);

  // 8) delta projection + double layernorm (ln then dn), reuse ysf for dproj
  cvt_f32_f16<<<cvtg, cvtb, 0, stream>>>(ysf, ysh, (long)NROWS * D_);
  gemm_wmma_bt<<<dim3((D_ + 255) / 256, NROWS / 16), 128, 0, stream>>>(ysh, dwoT, dproj, NROWS, D_, D_);
  layernorm_k<<<NROWS, 32, 0, stream>>>(dproj, dproj, ln_g, ln_b);
  layernorm_k<<<NROWS, 32, 0, stream>>>(dproj, dproj, dn_g, dn_b);

  // 9) gated combine -> out
  combine_gate<<<cvtg, cvtb, 0, stream>>>(gateb, flashproj, dproj, outp, (long)NROWS * D_);
}
